// Encoder_29686813950557
// MI455X (gfx1250) — compile-verified, matrix operand order checked by reading, not run on hardware
//
#include <hip/hip_runtime.h>
#include <hip/hip_bf16.h>

// ---------------------------------------------------------------------------
// Problem constants (from reference): VOCAB=32000, E=512, H=1024, L=2, B=64, T=256
// ---------------------------------------------------------------------------
#define VOCAB 32000
#define EDIM  512
#define HDIM  1024
#define GDIM  4096      // 4*H
#define BSZ   64
#define TLEN  256
#define ROWS  (TLEN*BSZ)  // 16384 time-parallel rows

typedef __bf16 bf16_t;
typedef __attribute__((ext_vector_type(16))) __bf16 v16bf;
typedef __attribute__((ext_vector_type(8)))  float  v8f;

__device__ __forceinline__ float sigm(float x) { return 1.0f / (1.0f + __expf(-x)); }

// ---------------------------------------------------------------------------
// Embedding gather + fp32->bf16 convert.  X0 row = t*B + b  (time-major).
// ---------------------------------------------------------------------------
__global__ void embed_gather_kernel(const int* __restrict__ src,
                                    const float* __restrict__ emb,
                                    bf16_t* __restrict__ X0) {
    const int row = blockIdx.x;           // t*B + b
    const int t = row / BSZ;
    const int b = row - t * BSZ;
    const int tok = src[b * TLEN + t];
    const float* __restrict__ srcrow = emb + (size_t)tok * EDIM;
    bf16_t* __restrict__ dstrow = X0 + (size_t)row * EDIM;
    int e = threadIdx.x;
    dstrow[e]       = (bf16_t)srcrow[e];
    dstrow[e + 256] = (bf16_t)srcrow[e + 256];
}

// ---------------------------------------------------------------------------
// LDS-tiled transpose + convert: WT[n*K + k] = (bf16) W[k*N + n].
// Weights stored [N,K] so WMMA B tiles are contiguous 32B per lane.
// ---------------------------------------------------------------------------
__global__ void transpose_conv_kernel(const float* __restrict__ W,
                                      bf16_t* __restrict__ WT,
                                      int K, int N) {
    __shared__ float tile[32][33];
    const int n0 = blockIdx.x * 32;
    const int k0 = blockIdx.y * 32;
    const int tx = threadIdx.x;   // 0..31
    const int ty = threadIdx.y;   // 0..7
#pragma unroll
    for (int i = 0; i < 4; ++i)
        tile[ty + i * 8][tx] = W[(size_t)(k0 + ty + i * 8) * N + n0 + tx];
    __syncthreads();
#pragma unroll
    for (int i = 0; i < 4; ++i)
        WT[(size_t)(n0 + ty + i * 8) * K + k0 + tx] = (bf16_t)tile[tx][ty + i * 8];
}

// ---------------------------------------------------------------------------
// Batched input projection: G[ROWS, GDIM] = X[ROWS, K] @ WT^T + bias.
// One wave computes a 16x64 tile (4 accumulators, A reused across 4 WMMAs).
// K loop is software-pipelined: k+32 tiles are loaded (and k+64 prefetched)
// while the k tiles feed the WMMA pipe.
// A layout: lanes 0-15 -> M rows, K 0..15 ; lanes 16-31 -> same rows, K 16..31.
// B layout: lanes 0-15 -> N cols, K 0..15 ; lanes 16-31 -> same cols, K 16..31.
// ---------------------------------------------------------------------------
__global__ void gemm_proj_kernel(const bf16_t* __restrict__ X,
                                 const bf16_t* __restrict__ WT,
                                 const float* __restrict__ bias,
                                 float* __restrict__ G,
                                 int K) {
    const int wid  = blockIdx.x * 8 + (threadIdx.x >> 5);
    const int lane = threadIdx.x & 31;
    const int lr   = lane & 15;
    const int half = lane >> 4;

    const int mt = wid >> 6;            // 0..ROWS/16-1
    const int ng = wid & 63;            // 0..63
    const int m0 = mt * 16;
    const int n0 = ng * 64;

    v8f acc[4];
#pragma unroll
    for (int s = 0; s < 4; ++s) {
        const float bv = bias[n0 + s * 16 + lr];
#pragma unroll
        for (int r = 0; r < 8; ++r) acc[s][r] = bv;
    }

    const bf16_t* __restrict__ arow = X + (size_t)(m0 + lr) * K + half * 16;
    const bf16_t* bptr[4];
#pragma unroll
    for (int s = 0; s < 4; ++s)
        bptr[s] = WT + (size_t)(n0 + s * 16 + lr) * K + half * 16;

    // ---- pipelined K loop: prologue ----
    v16bf a_cur = *(const v16bf*)(arow);
    v16bf b_cur[4];
#pragma unroll
    for (int s = 0; s < 4; ++s) b_cur[s] = *(const v16bf*)(bptr[s]);

    for (int k0 = 0; k0 < K - 32; k0 += 32) {
        // stage k0+32 while k0 computes; prefetch k0+64 into L2/L0
        const v16bf a_nxt = *(const v16bf*)(arow + k0 + 32);
        v16bf b_nxt[4];
#pragma unroll
        for (int s = 0; s < 4; ++s) {
            b_nxt[s] = *(const v16bf*)(bptr[s] + k0 + 32);
            __builtin_prefetch(bptr[s] + k0 + 64, 0, 3);
        }
#pragma unroll
        for (int s = 0; s < 4; ++s)
            acc[s] = __builtin_amdgcn_wmma_f32_16x16x32_bf16(
                false, a_cur, false, b_cur[s], (short)0, acc[s], false, false);
        a_cur = a_nxt;
#pragma unroll
        for (int s = 0; s < 4; ++s) b_cur[s] = b_nxt[s];
    }
    // ---- epilogue ----
#pragma unroll
    for (int s = 0; s < 4; ++s)
        acc[s] = __builtin_amdgcn_wmma_f32_16x16x32_bf16(
            false, a_cur, false, b_cur[s], (short)0, acc[s], false, false);

#pragma unroll
    for (int s = 0; s < 4; ++s) {
#pragma unroll
        for (int r = 0; r < 8; ++r) {
            const int m = m0 + r + half * 8;
            G[(size_t)m * GDIM + n0 + s * 16 + lr] = acc[s][r];
        }
    }
}

// ---------------------------------------------------------------------------
// One LSTM timestep.  Each wave owns a (16 rows x 16 H-cols) patch and computes
// the FOUR coupled gate tiles (i,f,n,o) so the c/h update stays in-register.
// Same double-buffered K pipeline as the projection GEMM (Wh streams from L2).
// G is pre-offset to timestep t (rows [B,4H]).  h is double-buffered.
// 256 waves total: grid = 32 blocks x 256 threads.
// ---------------------------------------------------------------------------
__global__ void lstm_step_kernel(const float* __restrict__ G,
                                 const bf16_t* __restrict__ hprev,   // [B,H]
                                 const bf16_t* __restrict__ WhT,     // [4H,H]
                                 bf16_t* __restrict__ hnext,         // [B,H]
                                 float*  __restrict__ c,             // [B,H]
                                 float*  __restrict__ hf,            // [B,H] fp32 state
                                 bf16_t* __restrict__ xout,          // X1 slice (layer0) or null
                                 float*  __restrict__ outMasked,     // d_out [B,T,H] (layer1) or null
                                 const int* __restrict__ lengths,
                                 int t) {
    const int wid  = blockIdx.x * 8 + (threadIdx.x >> 5);
    const int lane = threadIdx.x & 31;
    const int lr   = lane & 15;
    const int half = lane >> 4;

    const int m0 = (wid >> 6) * 16;     // batch-row tile (4 tiles)
    const int j0 = (wid & 63) * 16;     // H-column tile (64 tiles)

    // init accumulators from precomputed x@Wx + b
    v8f acc[4];
#pragma unroll
    for (int g = 0; g < 4; ++g) {
#pragma unroll
        for (int r = 0; r < 8; ++r) {
            const int m = m0 + r + half * 8;
            acc[g][r] = G[(size_t)m * GDIM + g * HDIM + j0 + lr];
        }
    }

    // recurrent GEMM: + h_prev @ Wh  (K = H = 1024, shared A across 4 gates)
    const bf16_t* __restrict__ arow = hprev + (size_t)(m0 + lr) * HDIM + half * 16;
    const bf16_t* bptr[4];
#pragma unroll
    for (int g = 0; g < 4; ++g)
        bptr[g] = WhT + (size_t)(g * HDIM + j0 + lr) * HDIM + half * 16;

    v16bf a_cur = *(const v16bf*)(arow);
    v16bf b_cur[4];
#pragma unroll
    for (int g = 0; g < 4; ++g) b_cur[g] = *(const v16bf*)(bptr[g]);

    for (int k0 = 0; k0 < HDIM - 32; k0 += 32) {
        const v16bf a_nxt = *(const v16bf*)(arow + k0 + 32);
        v16bf b_nxt[4];
#pragma unroll
        for (int g = 0; g < 4; ++g) {
            b_nxt[g] = *(const v16bf*)(bptr[g] + k0 + 32);
            __builtin_prefetch(bptr[g] + k0 + 64, 0, 3);
        }
#pragma unroll
        for (int g = 0; g < 4; ++g)
            acc[g] = __builtin_amdgcn_wmma_f32_16x16x32_bf16(
                false, a_cur, false, b_cur[g], (short)0, acc[g], false, false);
        a_cur = a_nxt;
#pragma unroll
        for (int g = 0; g < 4; ++g) b_cur[g] = b_nxt[g];
    }
#pragma unroll
    for (int g = 0; g < 4; ++g)
        acc[g] = __builtin_amdgcn_wmma_f32_16x16x32_bf16(
            false, a_cur, false, b_cur[g], (short)0, acc[g], false, false);

    // elementwise cell update (gate order i,f,n,o along 4H)
    const int j = j0 + lr;
#pragma unroll
    for (int r = 0; r < 8; ++r) {
        const int m = m0 + r + half * 8;
        const float gi = sigm(acc[0][r]);
        const float gf = sigm(acc[1][r]);
        const float gn = tanhf(acc[2][r]);
        const float go = sigm(acc[3][r]);
        const size_t idx = (size_t)m * HDIM + j;
        const float cn = gf * c[idx] + gi * gn;
        const float hn = go * tanhf(cn);
        c[idx]     = cn;
        hf[idx]    = hn;
        hnext[idx] = (bf16_t)hn;
        if (xout) xout[idx] = (bf16_t)hn;                       // feed layer-1 proj
        if (outMasked) {                                        // masked [B,T,H]
            const float v = (t < lengths[m]) ? hn : 0.0f;
            outMasked[((size_t)m * TLEN + t) * HDIM + j] = v;
        }
    }
}

// ---------------------------------------------------------------------------
// small helpers
// ---------------------------------------------------------------------------
__global__ void zero_u32_kernel(unsigned int* __restrict__ p, int n) {
    const int i = blockIdx.x * blockDim.x + threadIdx.x;
    if (i < n) p[i] = 0u;
}
__global__ void copy_f32_kernel(const float* __restrict__ s, float* __restrict__ d, int n) {
    const int i = blockIdx.x * blockDim.x + threadIdx.x;
    if (i < n) d[i] = s[i];
}

// ---------------------------------------------------------------------------
extern "C" void kernel_launch(void* const* d_in, const int* in_sizes, int n_in,
                              void* d_out, int out_size, void* d_ws, size_t ws_size,
                              hipStream_t stream) {
    const int*   src   = (const int*)  d_in[0];
    const int*   lens  = (const int*)  d_in[1];
    const float* emb   = (const float*)d_in[2];
    const float* Wx0   = (const float*)d_in[3];
    const float* Wh0   = (const float*)d_in[4];
    const float* b0    = (const float*)d_in[5];
    const float* Wx1   = (const float*)d_in[6];
    const float* Wh1   = (const float*)d_in[7];
    const float* b1    = (const float*)d_in[8];
    float* out = (float*)d_out;

    // ---- workspace carving (256B aligned) ----
    char* ws = (char*)d_ws;
    size_t off = 0;
    auto carve = [&](size_t bytes) -> char* {
        char* p = ws + off;
        off += (bytes + 255) & ~(size_t)255;
        return p;
    };
    bf16_t* X0    = (bf16_t*)carve((size_t)ROWS * EDIM * 2);
    bf16_t* X1    = (bf16_t*)carve((size_t)ROWS * HDIM * 2);
    float*  G     = (float*) carve((size_t)ROWS * GDIM * 4);
    bf16_t* WxT0  = (bf16_t*)carve((size_t)GDIM * EDIM * 2);
    bf16_t* WhT0  = (bf16_t*)carve((size_t)GDIM * HDIM * 2);
    bf16_t* WxT1  = (bf16_t*)carve((size_t)GDIM * HDIM * 2);
    bf16_t* WhT1  = (bf16_t*)carve((size_t)GDIM * HDIM * 2);
    bf16_t* h0a   = (bf16_t*)carve((size_t)BSZ * HDIM * 2);
    bf16_t* h0b   = (bf16_t*)carve((size_t)BSZ * HDIM * 2);
    bf16_t* h1a   = (bf16_t*)carve((size_t)BSZ * HDIM * 2);
    bf16_t* h1b   = (bf16_t*)carve((size_t)BSZ * HDIM * 2);
    float*  c0    = (float*) carve((size_t)BSZ * HDIM * 4);
    float*  c1    = (float*) carve((size_t)BSZ * HDIM * 4);
    float*  h0f   = (float*) carve((size_t)BSZ * HDIM * 4);
    float*  h1f   = (float*) carve((size_t)BSZ * HDIM * 4);

    // ---- zero initial recurrent state (h_0, c_0 for both layers) ----
    zero_u32_kernel<<<(BSZ * HDIM / 2 + 255) / 256, 256, 0, stream>>>((unsigned*)h0a, BSZ * HDIM / 2);
    zero_u32_kernel<<<(BSZ * HDIM / 2 + 255) / 256, 256, 0, stream>>>((unsigned*)h1a, BSZ * HDIM / 2);
    zero_u32_kernel<<<(BSZ * HDIM + 255) / 256, 256, 0, stream>>>((unsigned*)c0, BSZ * HDIM);
    zero_u32_kernel<<<(BSZ * HDIM + 255) / 256, 256, 0, stream>>>((unsigned*)c1, BSZ * HDIM);

    // ---- embedding gather -> bf16 (time-major rows) ----
    embed_gather_kernel<<<ROWS, 256, 0, stream>>>(src, emb, X0);

    // ---- weight convert + transpose to [N,K] bf16 ----
    {
        dim3 blk(32, 8);
        transpose_conv_kernel<<<dim3(GDIM / 32, EDIM / 32), blk, 0, stream>>>(Wx0, WxT0, EDIM, GDIM);
        transpose_conv_kernel<<<dim3(GDIM / 32, HDIM / 32), blk, 0, stream>>>(Wh0, WhT0, HDIM, GDIM);
        transpose_conv_kernel<<<dim3(GDIM / 32, HDIM / 32), blk, 0, stream>>>(Wx1, WxT1, HDIM, GDIM);
        transpose_conv_kernel<<<dim3(GDIM / 32, HDIM / 32), blk, 0, stream>>>(Wh1, WhT1, HDIM, GDIM);
    }

    const int projBlocks = (ROWS / 16) * (GDIM / 64) / 8;   // 8192

    // ================= layer 0 =================
    gemm_proj_kernel<<<projBlocks, 256, 0, stream>>>(X0, WxT0, b0, G, EDIM);
    for (int t = 0; t < TLEN; ++t) {
        const bf16_t* hp = (t & 1) ? h0b : h0a;
        bf16_t*       hn = (t & 1) ? h0a : h0b;
        lstm_step_kernel<<<32, 256, 0, stream>>>(
            G + (size_t)t * BSZ * GDIM, hp, WhT0, hn, c0, h0f,
            X1 + (size_t)t * BSZ * HDIM, nullptr, lens, t);
    }

    // ================= layer 1 =================
    gemm_proj_kernel<<<projBlocks, 256, 0, stream>>>(X1, WxT1, b1, G, HDIM);
    for (int t = 0; t < TLEN; ++t) {
        const bf16_t* hp = (t & 1) ? h1b : h1a;
        bf16_t*       hn = (t & 1) ? h1a : h1b;
        lstm_step_kernel<<<32, 256, 0, stream>>>(
            G + (size_t)t * BSZ * GDIM, hp, WhT1, hn, c1, h1f,
            nullptr, out, lens, t);
    }

    // ---- h_n [2,B,H] and c_n [2,B,H] after outputs [B,T,H] ----
    const size_t BTH = (size_t)BSZ * TLEN * HDIM;
    const int    BH  = BSZ * HDIM;
    copy_f32_kernel<<<(BH + 255) / 256, 256, 0, stream>>>(h0f, out + BTH,                 BH);
    copy_f32_kernel<<<(BH + 255) / 256, 256, 0, stream>>>(h1f, out + BTH + BH,            BH);
    copy_f32_kernel<<<(BH + 255) / 256, 256, 0, stream>>>(c0,  out + BTH + 2 * (size_t)BH, BH);
    copy_f32_kernel<<<(BH + 255) / 256, 256, 0, stream>>>(c1,  out + BTH + 3 * (size_t)BH, BH);
}
